// PPIGAT_20856361189534
// MI455X (gfx1250) — matrix operand order repaired and checked by reading, not compiled, not run on hardware
//
#include <hip/hip_runtime.h>

// ---------------- problem constants (match reference) ----------------
#define N_NODES   50000
#define N_EDGES   600000
#define N_PRED    200000
#define IN_CH     128
#define HIDC      128          // per-head channels (both layers)
#define H1        4
#define H2        2
#define F1        (H1*HIDC)    // 512
#define F2        (H2*HIDC)    // 256
#define EE        (N_EDGES + N_NODES)   // edges + self loops = 650000
#define BN_EPS    1e-5f
#define NEG_SLOPE 0.2f

typedef unsigned short u16;
typedef unsigned int   u32;
typedef __attribute__((ext_vector_type(16))) __bf16 bf16x16;
typedef __attribute__((ext_vector_type(8)))  float  f32x8;
typedef __attribute__((ext_vector_type(4)))  float  f32x4;
typedef __attribute__((ext_vector_type(4)))  u32    u32x4;
typedef __attribute__((ext_vector_type(4)))  u16    u16x4;

union Frag { bf16x16 v; u32x4 q[2]; };

// ---------------- helpers ----------------
__device__ __forceinline__ u16 f2bf(float f) {
  union { float f; u32 u; } x; x.f = f;
  u32 r = x.u + 0x7FFFu + ((x.u >> 16) & 1u);   // round-to-nearest-even
  return (u16)(r >> 16);
}
// monotonic float -> uint key (for atomicMax-based segment max)
__device__ __forceinline__ u32 fkey(float f) {
  union { float f; u32 u; } x; x.f = f;
  return (x.u & 0x80000000u) ? ~x.u : (x.u | 0x80000000u);
}
__device__ __forceinline__ float funkey(u32 k) {
  union { u32 u; float f; } x;
  x.u = (k & 0x80000000u) ? (k & 0x7FFFFFFFu) : ~k;
  return x.f;
}
__device__ __forceinline__ int edge_src(const int* ei, int e) {
  return e < N_EDGES ? ei[e] : e - N_EDGES;                 // self-loop tail
}
__device__ __forceinline__ int edge_dst(const int* ei, int e) {
  return e < N_EDGES ? ei[N_EDGES + e] : e - N_EDGES;
}

// ---------------- elementwise / setup kernels ----------------
__global__ void k_fill_u32v4(u32x4* __restrict__ p, u32 val, long n4) {
  long i = (long)blockIdx.x * blockDim.x + threadIdx.x;
  if (i < n4) { u32x4 v = { val, val, val, val }; p[i] = v; }
}

__global__ void k_f32_to_bf16(const float* __restrict__ src, u16* __restrict__ dst, long n) {
  long i = (long)blockIdx.x * blockDim.x + threadIdx.x;
  if (i < n) dst[i] = f2bf(src[i]);
}

// WT[n*K + k] = bf16(W[k*Ncol + n])   (weights -> column-major bf16)
__global__ void k_transpose_to_bf16(const float* __restrict__ W, u16* __restrict__ WT,
                                    int K, int Ncol) {
  long i = (long)blockIdx.x * blockDim.x + threadIdx.x;
  long total = (long)K * Ncol;
  if (i >= total) return;
  int n = (int)(i / K);
  int k = (int)(i % K);
  WT[i] = f2bf(W[(size_t)k * Ncol + n]);
}

// ---------------- WMMA GEMM: C[M,Nc] = A[M,K](bf16,row) @ B (BT: [Nc,K] bf16) + bias ----
// One wave computes one 16x16 output tile; M,Nc multiples of 16, K multiple of 32.
__global__ __launch_bounds__(256)
void k_gemm_bf16(const u16* __restrict__ A, const u16* __restrict__ BT,
                 const float* __restrict__ bias, float* __restrict__ C,
                 int M, int Nc, int K) {
  int lane = threadIdx.x & 31;
  long tile = (long)blockIdx.x * (blockDim.x >> 5) + (threadIdx.x >> 5);
  int tilesN = Nc >> 4;
  long tilesTotal = (long)(M >> 4) * tilesN;
  if (tile >= tilesTotal) return;
  int tm = (int)(tile / tilesN);
  int tn = (int)(tile % tilesN);

  int half = lane >> 4;               // 0: lanes 0-15, 1: lanes 16-31
  int l16  = lane & 15;
  // A fragment: lane holds row M = tm*16+l16; K chunks kbA..kbA+7 and kbA+16..kbA+23
  const u16* Arow = A + (size_t)(tm * 16 + l16) * K;
  int kbA = half * 8;
  // B fragment: lane holds col N = tn*16+l16; K chunks kbB..kbB+7 and kbB+16..kbB+23
  const u16* Bcol = BT + (size_t)(tn * 16 + l16) * K;
  int kbB = half * 8;

  f32x8 acc = {};
  for (int k = 0; k < K; k += 32) {
    Frag a, b;
    a.q[0] = *(const u32x4*)(Arow + k + kbA);
    a.q[1] = *(const u32x4*)(Arow + k + kbA + 16);
    b.q[0] = *(const u32x4*)(Bcol + k + kbB);
    b.q[1] = *(const u32x4*)(Bcol + k + kbB + 16);
    acc = __builtin_amdgcn_wmma_f32_16x16x32_bf16(
        false, a.v, false, b.v, (short)0, acc, false, false);
  }
  // C/D layout: VGPR r -> M = r (+8 for lanes 16-31), N = lane&15
  int outCol = tn * 16 + l16;
  int rowBase = tm * 16 + half * 8;
  float bc = bias[outCol];
#pragma unroll
  for (int r = 0; r < 8; ++r)
    C[(size_t)(rowBase + r) * Nc + outCol] = acc[r] + bc;
}

// ---------------- GATv2 edge kernels ----------------
// wave per (edge, head): lane owns one contiguous float4 of the 128 channels.
// score = sum_c att[h,c]*leakyrelu(xl[src,h,c]+xr[dst,h,c]); atomicMax per dst node.
__global__ __launch_bounds__(256)
void k_edge_scores(const float* __restrict__ xl, const float* __restrict__ xr,
                   const int* __restrict__ ei, const float* __restrict__ att,
                   float* __restrict__ score, u32* __restrict__ maxk, int H) {
  int lane = threadIdx.x & 31;
  long task = (long)blockIdx.x * (blockDim.x >> 5) + (threadIdx.x >> 5);
  if (task >= (long)EE * H) return;
  int e = (int)(task / H);
  int h = (int)(task % H);
  int src = edge_src(ei, e), dst = edge_dst(ei, e);
  int c = lane * 4;                   // 32 lanes x float4 == 128 channels
  const float* pl = xl + (size_t)src * H * HIDC + (size_t)h * HIDC + c;
  const float* pp = xr + (size_t)dst * H * HIDC + (size_t)h * HIDC + c;
  const float* pa = att + (size_t)h * HIDC + c;
  f32x4 l = *(const f32x4*)pl;
  f32x4 r = *(const f32x4*)pp;
  f32x4 a = *(const f32x4*)pa;
  float s = 0.f;
#pragma unroll
  for (int j = 0; j < 4; ++j) {
    float t = l[j] + r[j];
    t = t >= 0.f ? t : NEG_SLOPE * t;
    s += a[j] * t;
  }
#pragma unroll
  for (int m = 16; m; m >>= 1) s += __shfl_xor(s, m, 32);
  if (lane == 0) {
    score[task] = s;
    atomicMax(&maxk[(size_t)dst * H + h], fkey(s));
  }
}

// thread per (edge, head): ex = exp(score - max[dst]); denom[dst] += ex (score overwritten)
__global__ void k_edge_exp(float* __restrict__ score, const u32* __restrict__ maxk,
                           float* __restrict__ denom, const int* __restrict__ ei, int H) {
  long t = (long)blockIdx.x * blockDim.x + threadIdx.x;
  if (t >= (long)EE * H) return;
  int e = (int)(t / H);
  int h = (int)(t % H);
  int dst = edge_dst(ei, e);
  float ex = expf(score[t] - funkey(maxk[(size_t)dst * H + h]));
  score[t] = ex;
  atomicAdd(&denom[(size_t)dst * H + h], ex);
}

// wave per (edge, head): out[dst] += (ex/denom[dst]) * xl[src]; lane owns one float4.
__global__ __launch_bounds__(256)
void k_edge_aggregate(const float* __restrict__ ex, const float* __restrict__ denom,
                      const float* __restrict__ xl, const int* __restrict__ ei,
                      float* __restrict__ outp, int H) {
  int lane = threadIdx.x & 31;
  long task = (long)blockIdx.x * (blockDim.x >> 5) + (threadIdx.x >> 5);
  if (task >= (long)EE * H) return;
  int e = (int)(task / H);
  int h = (int)(task % H);
  int src = edge_src(ei, e), dst = edge_dst(ei, e);
  int c = lane * 4;
  const float* pl = xl + (size_t)src * H * HIDC + (size_t)h * HIDC + c;
  float* po = outp + (size_t)dst * H * HIDC + (size_t)h * HIDC + c;
  // prefetch gather source while the alpha divide dependency resolves
  __builtin_prefetch(pl, 0, 0);
  float alpha = ex[task] / denom[(size_t)dst * H + h];
  f32x4 l = *(const f32x4*)pl;
#pragma unroll
  for (int j = 0; j < 4; ++j)
    atomicAdd(&po[j], alpha * l[j]);
}

// ---------------- fused (out+bias) -> BN(eval) -> ReLU (vectorized x4) ----------------
__global__ void k_bn_relu_bf16(const float* __restrict__ acc, const float* __restrict__ bias,
                               const float* __restrict__ g, const float* __restrict__ b,
                               const float* __restrict__ m, const float* __restrict__ v,
                               u16* __restrict__ outp, long n4, int F) {
  long i4 = (long)blockIdx.x * blockDim.x + threadIdx.x;
  if (i4 >= n4) return;
  long i = i4 * 4;
  int f = (int)(i % F);
  f32x4 a = *(const f32x4*)(acc + i);
  f32x4 bi = *(const f32x4*)(bias + f);
  f32x4 gg = *(const f32x4*)(g + f);
  f32x4 bb = *(const f32x4*)(b + f);
  f32x4 mm = *(const f32x4*)(m + f);
  f32x4 vv = *(const f32x4*)(v + f);
  u16x4 o;
#pragma unroll
  for (int j = 0; j < 4; ++j) {
    float val = a[j] + bi[j];
    val = (val - mm[j]) * rsqrtf(vv[j] + BN_EPS) * gg[j] + bb[j];
    o[j] = f2bf(val > 0.f ? val : 0.f);
  }
  *(u16x4*)(outp + i) = o;
}

__global__ void k_bn_relu_f32(const float* __restrict__ acc, const float* __restrict__ bias,
                              const float* __restrict__ g, const float* __restrict__ b,
                              const float* __restrict__ m, const float* __restrict__ v,
                              float* __restrict__ outp, long n4, int F) {
  long i4 = (long)blockIdx.x * blockDim.x + threadIdx.x;
  if (i4 >= n4) return;
  long i = i4 * 4;
  int f = (int)(i % F);
  f32x4 a = *(const f32x4*)(acc + i);
  f32x4 bi = *(const f32x4*)(bias + f);
  f32x4 gg = *(const f32x4*)(g + f);
  f32x4 bb = *(const f32x4*)(b + f);
  f32x4 mm = *(const f32x4*)(m + f);
  f32x4 vv = *(const f32x4*)(v + f);
  f32x4 o;
#pragma unroll
  for (int j = 0; j < 4; ++j) {
    float val = a[j] + bi[j];
    val = (val - mm[j]) * rsqrtf(vv[j] + BN_EPS) * gg[j] + bb[j];
    o[j] = val > 0.f ? val : 0.f;
  }
  *(f32x4*)(outp + i) = o;
}

// ---------------- final link prediction head: wave per prediction edge ----------------
// F2=256 channels: each lane owns two float4 chunks of z[s] and z[d].
__global__ __launch_bounds__(256)
void k_predict(const float* __restrict__ z, const int* __restrict__ pe,
               const float* __restrict__ pr, const float* __restrict__ pert,
               const float* __restrict__ Wp, const float* __restrict__ bp,
               float* __restrict__ outp) {
  int lane = threadIdx.x & 31;
  long p = (long)blockIdx.x * (blockDim.x >> 5) + (threadIdx.x >> 5);
  if (p >= N_PRED) return;
  int s = pe[p], d = pe[N_PRED + p];
  const float* zs = z + (size_t)s * F2;
  const float* zd = z + (size_t)d * F2;
  float a0 = 0.f, a1 = 0.f;
#pragma unroll
  for (int base = 0; base < F2; base += 128) {
    int f = base + lane * 4;
    f32x4 vs = *(const f32x4*)(zs + f);
    f32x4 vd = *(const f32x4*)(zd + f);
#pragma unroll
    for (int j = 0; j < 4; ++j) {
      a0 += vs[j] * Wp[(f + j) * 2 + 0] + vd[j] * Wp[(F2 + f + j) * 2 + 0];
      a1 += vs[j] * Wp[(f + j) * 2 + 1] + vd[j] * Wp[(F2 + f + j) * 2 + 1];
    }
  }
#pragma unroll
  for (int m = 16; m; m >>= 1) { a0 += __shfl_xor(a0, m, 32); a1 += __shfl_xor(a1, m, 32); }
  if (lane == 0) {
    float prs = pr[s], prd = pr[d], pp = pert[s] * pert[d];
    a0 += prs * Wp[(2*F2+0)*2+0] + prd * Wp[(2*F2+1)*2+0] + pp * Wp[(2*F2+2)*2+0] + bp[0];
    a1 += prs * Wp[(2*F2+0)*2+1] + prd * Wp[(2*F2+1)*2+1] + pp * Wp[(2*F2+2)*2+1] + bp[1];
    outp[p * 2 + 0] = a0;
    outp[p * 2 + 1] = a1;
  }
}

// ---------------- host orchestration ----------------
static inline int cdiv(long a, long b) { return (int)((a + b - 1) / b); }

extern "C" void kernel_launch(void* const* d_in, const int* in_sizes, int n_in,
                              void* d_out, int out_size, void* d_ws, size_t ws_size,
                              hipStream_t stream) {
  (void)in_sizes; (void)n_in; (void)out_size; (void)ws_size;
  const float* x     = (const float*)d_in[0];
  const int*   ei    = (const int*)d_in[1];
  const int*   pe    = (const int*)d_in[2];
  const float* pert  = (const float*)d_in[3];
  const float* pr    = (const float*)d_in[4];
  const float* Wl1   = (const float*)d_in[5];
  const float* bl1   = (const float*)d_in[6];
  const float* Wr1   = (const float*)d_in[7];
  const float* br1   = (const float*)d_in[8];
  const float* att1  = (const float*)d_in[9];
  const float* bias1 = (const float*)d_in[10];
  const float* bn1g  = (const float*)d_in[11];
  const float* bn1b  = (const float*)d_in[12];
  const float* bn1m  = (const float*)d_in[13];
  const float* bn1v  = (const float*)d_in[14];
  const float* Wl2   = (const float*)d_in[15];
  const float* bl2   = (const float*)d_in[16];
  const float* Wr2   = (const float*)d_in[17];
  const float* br2   = (const float*)d_in[18];
  const float* att2  = (const float*)d_in[19];
  const float* bias2 = (const float*)d_in[20];
  const float* bn2g  = (const float*)d_in[21];
  const float* bn2b  = (const float*)d_in[22];
  const float* bn2m  = (const float*)d_in[23];
  const float* bn2v  = (const float*)d_in[24];
  const float* Wp    = (const float*)d_in[25];
  const float* bp    = (const float*)d_in[26];
  float* out = (float*)d_out;

  // bump allocator over d_ws (256B aligned regions)
  char* ws = (char*)d_ws;
  size_t off = 0;
  auto alloc = [&](size_t bytes) -> char* {
    char* p = ws + off;
    off = (off + bytes + 255) & ~(size_t)255;
    return p;
  };
  u16*   xbf   = (u16*)  alloc((size_t)N_NODES * IN_CH * 2);   // x in bf16
  u16*   wlt1  = (u16*)  alloc((size_t)F1 * IN_CH * 2);        // Wl1^T bf16
  u16*   wrt1  = (u16*)  alloc((size_t)F1 * IN_CH * 2);
  u16*   wlt2  = (u16*)  alloc((size_t)F2 * F1 * 2);
  u16*   wrt2  = (u16*)  alloc((size_t)F2 * F1 * 2);
  float* xl    = (float*)alloc((size_t)N_NODES * F1 * 4);      // layer1 xl / layer2 xl
  float* xr    = (float*)alloc((size_t)N_NODES * F1 * 4);      // layer1 xr / layer2 xr
  float* agg   = (float*)alloc((size_t)N_NODES * F1 * 4);      // L1 acc; later L2 acc + z
  u16*   h1bf  = (u16*)  alloc((size_t)N_NODES * F1 * 2);      // relu(bn1(...)) in bf16
  float* score = (float*)alloc((size_t)EE * H1 * 4);           // scores then exp
  u32*   maxk  = (u32*)  alloc((size_t)(N_NODES * H1 + 64) * 4);
  float* denom = (float*)alloc((size_t)(N_NODES * H1 + 64) * 4);
  float* agg2  = agg;                                          // N*F2 f32 (first half)
  float* zbuf  = agg + (size_t)N_NODES * F2;                   // N*F2 f32 (second half)

  // ---- precision conversion + weight packing ----
  k_f32_to_bf16<<<cdiv((long)N_NODES*IN_CH,256),256,0,stream>>>(x, xbf, (long)N_NODES*IN_CH);
  k_transpose_to_bf16<<<cdiv((long)IN_CH*F1,256),256,0,stream>>>(Wl1, wlt1, IN_CH, F1);
  k_transpose_to_bf16<<<cdiv((long)IN_CH*F1,256),256,0,stream>>>(Wr1, wrt1, IN_CH, F1);
  k_transpose_to_bf16<<<cdiv((long)F1*F2,256),256,0,stream>>>(Wl2, wlt2, F1, F2);
  k_transpose_to_bf16<<<cdiv((long)F1*F2,256),256,0,stream>>>(Wr2, wrt2, F1, F2);

  // ---- layer 1: GEMMs (WMMA bf16) ----
  {
    long tiles = ((long)N_NODES/16) * (F1/16);
    k_gemm_bf16<<<cdiv(tiles,8),256,0,stream>>>(xbf, wlt1, bl1, xl, N_NODES, F1, IN_CH);
    k_gemm_bf16<<<cdiv(tiles,8),256,0,stream>>>(xbf, wrt1, br1, xr, N_NODES, F1, IN_CH);
  }
  // ---- layer 1: segment softmax attention ----
  k_fill_u32v4<<<cdiv((long)N_NODES*H1/4,256),256,0,stream>>>((u32x4*)maxk, 0u, (long)N_NODES*H1/4);
  k_fill_u32v4<<<cdiv((long)N_NODES*H1/4,256),256,0,stream>>>((u32x4*)denom, 0u, (long)N_NODES*H1/4);
  k_fill_u32v4<<<cdiv((long)N_NODES*F1/4,256),256,0,stream>>>((u32x4*)agg, 0u, (long)N_NODES*F1/4);
  k_edge_scores<<<cdiv((long)EE*H1,8),256,0,stream>>>(xl, xr, ei, att1, score, maxk, H1);
  k_edge_exp<<<cdiv((long)EE*H1,256),256,0,stream>>>(score, maxk, denom, ei, H1);
  k_edge_aggregate<<<cdiv((long)EE*H1,8),256,0,stream>>>(score, denom, xl, ei, agg, H1);
  k_bn_relu_bf16<<<cdiv((long)N_NODES*F1/4,256),256,0,stream>>>(
      agg, bias1, bn1g, bn1b, bn1m, bn1v, h1bf, (long)N_NODES*F1/4, F1);

  // ---- layer 2: GEMMs (WMMA bf16) ----
  {
    long tiles = ((long)N_NODES/16) * (F2/16);
    k_gemm_bf16<<<cdiv(tiles,8),256,0,stream>>>(h1bf, wlt2, bl2, xl, N_NODES, F2, F1);
    k_gemm_bf16<<<cdiv(tiles,8),256,0,stream>>>(h1bf, wrt2, br2, xr, N_NODES, F2, F1);
  }
  // ---- layer 2: segment softmax attention ----
  k_fill_u32v4<<<cdiv((long)N_NODES*H2/4,256),256,0,stream>>>((u32x4*)maxk, 0u, (long)N_NODES*H2/4);
  k_fill_u32v4<<<cdiv((long)N_NODES*H2/4,256),256,0,stream>>>((u32x4*)denom, 0u, (long)N_NODES*H2/4);
  k_fill_u32v4<<<cdiv((long)N_NODES*F2/4,256),256,0,stream>>>((u32x4*)agg2, 0u, (long)N_NODES*F2/4);
  k_edge_scores<<<cdiv((long)EE*H2,8),256,0,stream>>>(xl, xr, ei, att2, score, maxk, H2);
  k_edge_exp<<<cdiv((long)EE*H2,256),256,0,stream>>>(score, maxk, denom, ei, H2);
  k_edge_aggregate<<<cdiv((long)EE*H2,8),256,0,stream>>>(score, denom, xl, ei, agg2, H2);
  k_bn_relu_f32<<<cdiv((long)N_NODES*F2/4,256),256,0,stream>>>(
      agg2, bias2, bn2g, bn2b, bn2m, bn2v, zbuf, (long)N_NODES*F2/4, F2);

  // ---- link prediction head ----
  k_predict<<<cdiv((long)N_PRED,8),256,0,stream>>>(zbuf, pe, pr, pert, Wp, bp, out);
}
